// VerticalAttention_34222299415090
// MI455X (gfx1250) — matrix operand order repaired and checked by reading, not compiled, not run on hardware
//
#include <hip/hip_runtime.h>
#include <hip/hip_bf16.h>

// ---------------------------------------------------------------------------
// VerticalAttention on MI455X (gfx1250, wave32, WMMA)
//
// Roofline: features (16,256,128,512) fp32 = 1.07GB read twice (pool pass +
// context pass) -> ~92us at 23.3 TB/s; all GEMMs are <1 GFLOP, so they run on
// v_wmma_f32_16x16x32_f16 essentially for free. Streaming kernels use b128
// loads; WMMA operands are pre-transposed f16 in workspace so fragments load
// as contiguous v8h (16B) chunks matching ISA 7.12.2 layouts.
// ---------------------------------------------------------------------------

#define B_    16
#define C_    256
#define H_    128
#define W_    512
#define MINW_ 128
#define MINH_ 32
#define AF_   256
#define HS_   256

typedef __attribute__((ext_vector_type(16))) _Float16 v16h;
typedef __attribute__((ext_vector_type(8)))  _Float16 v8h;
typedef __attribute__((ext_vector_type(8)))  float    v8f;

// ---- WMMA fragment loaders (wave32, 16-bit operands) ----------------------
// A (16x32 f16) per ISA 7.12.2: lane l, row M=l&15, h=l>>4:
//   element e -> K = 8*h + e + (e>=8 ? 8 : 0)   => two contiguous 8-half runs
__device__ __forceinline__ v16h load_a_frag(const _Float16* __restrict__ row,
                                            int k0, int hi) {
    v8h lo = *(const v8h*)(row + k0 + 8 * hi);
    v8h hv = *(const v8h*)(row + k0 + 16 + 8 * hi);
    v16h r;
#pragma unroll
    for (int i = 0; i < 8; ++i) { r[i] = lo[i]; r[8 + i] = hv[i]; }
    return r;
}
// B (32x16 f16), stored K-contiguous per output column (we pre-transpose the
// weights): lane l, col N=l&15, h=l>>4: element e -> K = 8*h + e + (e>=8 ? 8 : 0)
__device__ __forceinline__ v16h load_b_frag(const _Float16* __restrict__ col,
                                            int k0, int hi) {
    v8h lo = *(const v8h*)(col + k0 + 8 * hi);
    v8h hv = *(const v8h*)(col + k0 + 16 + 8 * hi);
    v16h r;
#pragma unroll
    for (int i = 0; i < 8; ++i) { r[i] = lo[i]; r[8 + i] = hv[i]; }
    return r;
}

// ---- Prep: weight transposes / f16 conversion -----------------------------
__global__ void k_prep_encT(const float* __restrict__ enc_w, _Float16* __restrict__ encT) {
    int idx = blockIdx.x * 256 + threadIdx.x;          // C_*AF_
    int c = idx / AF_, f = idx % AF_;
    encT[f * C_ + c] = (_Float16)enc_w[idx];
}
__global__ void k_prep_dcbT(const float* __restrict__ dcb_w, _Float16* __restrict__ dcbT) {
    int idx = blockIdx.x * 256 + threadIdx.x;          // AF_*32 (K padded 16->32)
    int f = idx >> 5, k = idx & 31;
    dcbT[idx] = (_Float16)(k < 16 ? dcb_w[k * AF_ + f] : 0.f);
}
__global__ void k_prep_convdT(const float* __restrict__ convd_w, _Float16* __restrict__ convdT) {
    int idx = blockIdx.x * 256 + threadIdx.x;          // 5*AF_*AF_  [t][f][g]
    int t = idx / (AF_ * AF_);
    int rem = idx % (AF_ * AF_);
    int f = rem / AF_, g = rem % AF_;
    convdT[idx] = (_Float16)convd_w[(f * AF_ + g) * 5 + t];
}

// ---- K1: fused adaptive max-pool (W 512->128) + dot(dw_w) -----------------
// One wave per (b,c,h) row of 512 floats (2KB): b128 loads, wave32 reduce.
// Output stored transposed+f16 as A-matrix for the s GEMM: hfeatT[b][h][c].
__global__ void k1_pool_dot(const float* __restrict__ feat,
                            const float* __restrict__ dw_w,
                            const float* __restrict__ dw_b,
                            _Float16* __restrict__ hfeatT) {
    int wave = (blockIdx.x * blockDim.x + threadIdx.x) >> 5;
    int lane = threadIdx.x & 31;
    int b = wave / (C_ * H_);
    int rem = wave % (C_ * H_);
    int c = rem / H_, h = rem % H_;
    const float4* src = (const float4*)(feat + (size_t)wave * W_) + lane * 4;
    float acc = 0.f;
#pragma unroll
    for (int j = 0; j < 4; ++j) {
        float4 v = src[j];                              // window of 4 along W
        float m = fmaxf(fmaxf(v.x, v.y), fmaxf(v.z, v.w));
        acc = fmaf(m, dw_w[lane * 4 + j], acc);
    }
#pragma unroll
    for (int off = 16; off > 0; off >>= 1) acc += __shfl_down(acc, off, 32);
    if (lane == 0)
        hfeatT[((size_t)b * H_ + h) * C_ + c] = (_Float16)(acc + dw_b[0]);
}

// ---- K2a: instance-norm(cat) + conv1d(k=15,pad=7) -> cbT (K padded to 32) -
__device__ __forceinline__ float blk_sum128(float v, float* red, int h) {
    red[h] = v; __syncthreads();
#pragma unroll
    for (int s = 64; s > 0; s >>= 1) { if (h < s) red[h] += red[h + s]; __syncthreads(); }
    float r = red[0]; __syncthreads();
    return r;
}
__global__ void k2a_cb(const float* __restrict__ paw, const float* __restrict__ cov,
                       const float* __restrict__ conv_w, const float* __restrict__ conv_b,
                       _Float16* __restrict__ cbT) {
    int b = blockIdx.x, h = threadIdx.x;               // 128 threads
    __shared__ float red[128];
    __shared__ float c0[H_ + 14], c1[H_ + 14];
    float x0 = paw[b * H_ + h];
    float x1 = fminf(fmaxf(cov[b * H_ + h], 0.f), 1.f);
    float s0  = blk_sum128(x0, red, h)       * (1.f / H_);
    float q0  = blk_sum128(x0 * x0, red, h)  * (1.f / H_);
    float s1  = blk_sum128(x1, red, h)       * (1.f / H_);
    float q1  = blk_sum128(x1 * x1, red, h)  * (1.f / H_);
    float r0 = rsqrtf(q0 - s0 * s0 + 1e-5f);
    float r1 = rsqrtf(q1 - s1 * s1 + 1e-5f);
    if (h < 7) { c0[h] = 0.f; c1[h] = 0.f; c0[135 + h] = 0.f; c1[135 + h] = 0.f; }
    c0[7 + h] = (x0 - s0) * r0;
    c1[7 + h] = (x1 - s1) * r1;
    __syncthreads();
    _Float16* out = cbT + ((size_t)b * H_ + h) * 32;
#pragma unroll
    for (int k = 0; k < 16; ++k) {
        float acc = conv_b[k];
#pragma unroll
        for (int t = 0; t < 15; ++t) {
            acc = fmaf(c0[h + t], conv_w[(k * 2 + 0) * 15 + t], acc);
            acc = fmaf(c1[h + t], conv_w[(k * 2 + 1) * 15 + t], acc);
        }
        out[k] = (_Float16)acc;
    }
#pragma unroll
    for (int k = 16; k < 32; ++k) out[k] = (_Float16)0.f;
}

// ---- K2b: per-batch broadcast vector: hh@hid_w + hid_b + enc_b + dcb_b ----
__global__ void k2b_hidbias(const float* __restrict__ hidden, const float* __restrict__ hid_w,
                            const float* __restrict__ hid_b, const float* __restrict__ enc_b,
                            const float* __restrict__ dcb_b, float* __restrict__ hidbias) {
    int b = blockIdx.x, f = threadIdx.x;               // 256 threads
    const float* hh = hidden + (size_t)b * HS_;        // hidden[0,0,b,:]
    float acc = hid_b[f] + enc_b[f] + dcb_b[f];
    for (int i = 0; i < HS_; ++i) acc = fmaf(hh[i], hid_w[i * AF_ + f], acc);
    hidbias[b * AF_ + f] = acc;
}

// ---- K3: s = tanh(hfeat^T @ enc_w + cb^T @ dcb_w + hidbias)  [WMMA] -------
// One wave per 16x16 tile; K=256 in 8 WMMA steps + 1 WMMA for the cb term.
__global__ void k3_s_wmma(const _Float16* __restrict__ hfeatT, const _Float16* __restrict__ encT,
                          const _Float16* __restrict__ cbT, const _Float16* __restrict__ dcbT,
                          const float* __restrict__ hidbias,
                          float* __restrict__ s32, _Float16* __restrict__ s16) {
    int tn = blockIdx.x, tm = blockIdx.y, b = blockIdx.z;
    int l = threadIdx.x, r = l & 15, hi = l >> 4;
    const _Float16* Arow = hfeatT + ((size_t)b * H_ + tm * 16 + r) * C_;
    const _Float16* Bcol = encT + (size_t)(tn * 16 + r) * C_;
    v8f acc = {};
#pragma unroll
    for (int k0 = 0; k0 < C_; k0 += 32) {
        v16h a  = load_a_frag(Arow, k0, hi);
        v16h bb = load_b_frag(Bcol, k0, hi);
        acc = __builtin_amdgcn_wmma_f32_16x16x32_f16(false, a, false, bb,
                                                     (short)0, acc, false, false);
    }
    {   // coverage-conv term, K padded 16->32
        const _Float16* A2 = cbT + ((size_t)b * H_ + tm * 16 + r) * 32;
        const _Float16* B2 = dcbT + (size_t)(tn * 16 + r) * 32;
        v16h a  = load_a_frag(A2, 0, hi);
        v16h bb = load_b_frag(B2, 0, hi);
        acc = __builtin_amdgcn_wmma_f32_16x16x32_f16(false, a, false, bb,
                                                     (short)0, acc, false, false);
    }
    int n = tn * 16 + r;
    float hb = hidbias[b * AF_ + n];
#pragma unroll
    for (int v = 0; v < 8; ++v) {
        int m = tm * 16 + v + 8 * hi;                  // C/D layout: VGPR v, half hi
        float sv = tanhf(acc[v] + hb);
        size_t idx = ((size_t)b * H_ + m) * AF_ + n;
        s32[idx] = sv;
        s16[idx] = (_Float16)sv;
    }
}

// ---- K4: align = s @ align_w + b; softmax over H ---------------------------
__global__ void k4_align_softmax(const float* __restrict__ s, const float* __restrict__ align_w,
                                 const float* __restrict__ align_b,
                                 float* __restrict__ attn_out, float* __restrict__ attn_ws) {
    int b = blockIdx.x, h = threadIdx.x;               // 128 threads
    const float4* row = (const float4*)(s + ((size_t)b * H_ + h) * AF_);
    const float4* wv = (const float4*)align_w;
    float acc = 0.f;
#pragma unroll
    for (int i = 0; i < AF_ / 4; ++i) {
        float4 v = row[i], w = wv[i];
        acc += v.x * w.x + v.y * w.y + v.z * w.z + v.w * w.w;
    }
    float a = acc + align_b[0];
    __shared__ float red[128];
    red[h] = a; __syncthreads();
#pragma unroll
    for (int s2 = 64; s2 > 0; s2 >>= 1) { if (h < s2) red[h] = fmaxf(red[h], red[h + s2]); __syncthreads(); }
    float amax = red[0]; __syncthreads();
    float e = __expf(a - amax);
    red[h] = e; __syncthreads();
#pragma unroll
    for (int s2 = 64; s2 > 0; s2 >>= 1) { if (h < s2) red[h] += red[h + s2]; __syncthreads(); }
    float w = e / red[0];
    attn_out[b * H_ + h] = w;
    attn_ws[b * H_ + h] = w;
}

// ---- K5: context[b,c,w] = sum_h features[b,c,h,w]*attn[b,h]  (2nd HBM pass)
__global__ void k5_context(const float* __restrict__ feat, const float* __restrict__ attn,
                           float* __restrict__ ctx) {
    int bc = blockIdx.x;                               // B_*C_ blocks, 128 threads
    int b = bc / C_;
    __shared__ float a_s[H_];
    if (threadIdx.x < H_) a_s[threadIdx.x] = attn[b * H_ + threadIdx.x];
    __syncthreads();
    const float4* src = (const float4*)(feat + (size_t)bc * H_ * W_);
    int t = threadIdx.x;
    float4 acc = {0.f, 0.f, 0.f, 0.f};
    for (int h = 0; h < H_; ++h) {
        if (h + 8 < H_) __builtin_prefetch(src + (h + 8) * (W_ / 4) + t, 0, 1);
        float4 v = src[h * (W_ / 4) + t];
        float a = a_s[h];
        acc.x = fmaf(v.x, a, acc.x); acc.y = fmaf(v.y, a, acc.y);
        acc.z = fmaf(v.z, a, acc.z); acc.w = fmaf(v.w, a, acc.w);
    }
    ((float4*)(ctx + (size_t)bc * W_))[t] = acc;
}

// ---- K6: convd as 5 shifted GEMMs [WMMA] + fused relu + maxpool(4) --------
__global__ void k6_convd_wmma(const _Float16* __restrict__ s16,
                              const _Float16* __restrict__ convdT,
                              const float* __restrict__ convd_b,
                              float* __restrict__ dpool) {
    int tn = blockIdx.x, tm = blockIdx.y, b = blockIdx.z;
    int l = threadIdx.x, r = l & 15, hi = l >> 4;
    v8f acc = {};
#pragma unroll
    for (int t = 0; t < 5; ++t) {
        int row = tm * 16 + r + t - 2;
        bool ok = (row >= 0) && (row < H_);
        const _Float16* Arow = s16 + ((size_t)b * H_ + (ok ? row : 0)) * AF_;
        const _Float16* Bcol = convdT + (size_t)t * AF_ * AF_ + (size_t)(tn * 16 + r) * AF_;
#pragma unroll
        for (int k0 = 0; k0 < AF_; k0 += 32) {
            v16h a = {};
            if (ok) a = load_a_frag(Arow, k0, hi);
            v16h bb = load_b_frag(Bcol, k0, hi);
            acc = __builtin_amdgcn_wmma_f32_16x16x32_f16(false, a, false, bb,
                                                         (short)0, acc, false, false);
        }
    }
    int f = tn * 16 + r;
    float cb = convd_b[f];
    // relu then maxpool(4): rows v=0..3 / v=4..7 (lane half hi shifts by 8 rows
    // = exactly two pool windows), identity 0 implements the relu clamp.
    float p0 = 0.f, p1 = 0.f;
#pragma unroll
    for (int v = 0; v < 4; ++v) p0 = fmaxf(p0, acc[v] + cb);
#pragma unroll
    for (int v = 4; v < 8; ++v) p1 = fmaxf(p1, acc[v] + cb);
    int mh = tm * 4 + 2 * hi;
    dpool[((size_t)b * AF_ + f) * MINH_ + mh]     = p0;
    dpool[((size_t)b * AF_ + f) * MINH_ + mh + 1] = p1;
}

// ---- K7: decision head -----------------------------------------------------
__global__ void k7_decision(const float* __restrict__ dpool, const float* __restrict__ dh_w,
                            const float* __restrict__ dh_b, const float* __restrict__ hidden,
                            const float* __restrict__ dec_w, const float* __restrict__ dec_b,
                            float* __restrict__ dec_out) {
    int b = blockIdx.x, f = threadIdx.x;               // 256 threads
    const float* dp = dpool + ((size_t)b * AF_ + f) * MINH_;
    float acc = dh_b[0];
#pragma unroll
    for (int i = 0; i < MINH_; ++i) acc = fmaf(dp[i], dh_w[i], acc);
    __shared__ float dd[AF_];
    dd[f] = fmaxf(acc, 0.f);
    __syncthreads();
    if (f < 2) {
        const float* hh = hidden + (size_t)b * HS_;
        float d = dec_b[f];
        for (int i = 0; i < HS_; ++i) d = fmaf(hh[i], dec_w[i * 2 + f], d);
        for (int i = 0; i < AF_; ++i) d = fmaf(dd[i], dec_w[(HS_ + i) * 2 + f], d);
        dec_out[b * 2 + f] = d;
    }
}

// ---------------------------------------------------------------------------
extern "C" void kernel_launch(void* const* d_in, const int* in_sizes, int n_in,
                              void* d_out, int out_size, void* d_ws, size_t ws_size,
                              hipStream_t stream) {
    const float* features = (const float*)d_in[0];
    const float* paw      = (const float*)d_in[1];
    const float* cov      = (const float*)d_in[2];
    const float* hidden   = (const float*)d_in[3];
    const float* dw_w     = (const float*)d_in[4];
    const float* dw_b     = (const float*)d_in[5];
    const float* enc_w    = (const float*)d_in[6];
    const float* enc_b    = (const float*)d_in[7];
    const float* conv_w   = (const float*)d_in[8];
    const float* conv_b   = (const float*)d_in[9];
    const float* dcb_w    = (const float*)d_in[10];
    const float* dcb_b    = (const float*)d_in[11];
    const float* hid_w    = (const float*)d_in[12];
    const float* hid_b    = (const float*)d_in[13];
    const float* align_w  = (const float*)d_in[14];
    const float* align_b  = (const float*)d_in[15];
    const float* convd_w  = (const float*)d_in[16];
    const float* convd_b  = (const float*)d_in[17];
    const float* dh_w     = (const float*)d_in[18];
    const float* dh_b     = (const float*)d_in[19];
    const float* dec_w    = (const float*)d_in[20];
    const float* dec_b    = (const float*)d_in[21];

    // workspace carve-out (256B aligned)
    char* ws = (char*)d_ws;
    size_t off = 0;
    auto carve = [&](size_t bytes) {
        void* p = ws + off;
        off = (off + bytes + 255) & ~(size_t)255;
        return p;
    };
    _Float16* hfeatT  = (_Float16*)carve((size_t)B_ * H_ * C_ * 2);      // 1 MB
    _Float16* encT    = (_Float16*)carve((size_t)C_ * AF_ * 2);          // 128 KB
    _Float16* cbT     = (_Float16*)carve((size_t)B_ * H_ * 32 * 2);      // 128 KB
    _Float16* dcbT    = (_Float16*)carve((size_t)AF_ * 32 * 2);          // 16 KB
    float*    hidbias = (float*)carve((size_t)B_ * AF_ * 4);             // 16 KB
    float*    s32     = (float*)carve((size_t)B_ * H_ * AF_ * 4);        // 2 MB
    _Float16* s16     = (_Float16*)carve((size_t)B_ * H_ * AF_ * 2);     // 1 MB
    _Float16* convdT  = (_Float16*)carve((size_t)5 * AF_ * AF_ * 2);     // 640 KB
    float*    attn_ws = (float*)carve((size_t)B_ * H_ * 4);              // 8 KB
    float*    dpool   = (float*)carve((size_t)B_ * AF_ * MINH_ * 4);     // 512 KB

    float* out_ctx  = (float*)d_out;                    // (B,C,W)
    float* out_attn = out_ctx + (size_t)B_ * C_ * W_;   // (B,H)
    float* out_dec  = out_attn + (size_t)B_ * H_;       // (B,2)

    // weight prep (tiny)
    k_prep_encT<<<(C_ * AF_) / 256, 256, 0, stream>>>(enc_w, encT);
    k_prep_dcbT<<<(AF_ * 32) / 256, 256, 0, stream>>>(dcb_w, dcbT);
    k_prep_convdT<<<(5 * AF_ * AF_) / 256, 256, 0, stream>>>(convd_w, convdT);

    // pass 1 over features: fused maxpool + dw dot (bandwidth bound)
    k1_pool_dot<<<(B_ * C_ * H_) / 8, 256, 0, stream>>>(features, dw_w, dw_b, hfeatT);

    // tiny side computations
    k2a_cb<<<B_, 128, 0, stream>>>(paw, cov, conv_w, conv_b, cbT);
    k2b_hidbias<<<B_, 256, 0, stream>>>(hidden, hid_w, hid_b, enc_b, dcb_b, hidbias);

    // s = tanh(GEMM + GEMM + broadcast)  [WMMA]
    k3_s_wmma<<<dim3(AF_ / 16, H_ / 16, B_), 32, 0, stream>>>(
        hfeatT, encT, cbT, dcbT, hidbias, s32, s16);

    // softmax attention
    k4_align_softmax<<<B_, 128, 0, stream>>>(s32, align_w, align_b, out_attn, attn_ws);

    // pass 2 over features: weighted context (bandwidth bound)
    k5_context<<<B_ * C_, 128, 0, stream>>>(features, attn_ws, out_ctx);

    // stop-decision conv as 5 shifted GEMMs [WMMA] with fused relu+pool
    k6_convd_wmma<<<dim3(AF_ / 16, H_ / 16, B_), 32, 0, stream>>>(
        s16, convdT, convd_b, dpool);

    // decision head
    k7_decision<<<B_, 256, 0, stream>>>(dpool, dh_w, dh_b, hidden, dec_w, dec_b, out_dec);
}
